// SuperEncoder_35759897706995
// MI455X (gfx1250) — matrix operand order repaired and checked by reading, not compiled, not run on hardware
//
#include <hip/hip_runtime.h>

typedef __attribute__((ext_vector_type(16))) _Float16 v16h;
typedef __attribute__((ext_vector_type(8)))  float    v8f;

#define THREADS 256

// 16-bit A-matrix 16x32 K mapping (ISA 7.12.2): VGPR pair p, lane-half h
__device__ __forceinline__ int kmapA(int p, int half) {
  return ((p < 4) ? (2 * p) : (2 * p + 8)) + 8 * half;
}
// 16-bit B-matrix 32x16 K mapping (same K order as A)
__device__ __forceinline__ int kmapB(int p, int half) {
  return ((p < 4) ? (2 * p) : (2 * p + 8)) + 8 * half;
}

__global__ __launch_bounds__(THREADS)
void k_deg_init(float* __restrict__ deg, int N) {
  int i = blockIdx.x * THREADS + threadIdx.x;
  if (i < N) deg[i] = 1.0f;                    // self-loop contributes 1
}

__global__ __launch_bounds__(THREADS)
void k_deg_count(const int* __restrict__ src, float* __restrict__ deg, int E) {
  int e = blockIdx.x * THREADS + threadIdx.x;
  if (e < E) atomicAdd(&deg[src[e]], 1.0f);
}

__global__ __launch_bounds__(THREADS)
void k_dis(float* __restrict__ deg, int N) {
  int i = blockIdx.x * THREADS + threadIdx.x;
  if (i < N) deg[i] = rsqrtf(deg[i]);          // deg >= 1 always
}

__global__ __launch_bounds__(THREADS)
void k_norm(const int* __restrict__ src, const int* __restrict__ dst,
            const float* __restrict__ dis, float* __restrict__ norm, int E) {
  int e = blockIdx.x * THREADS + threadIdx.x;
  if (e < E) norm[e] = dis[src[e]] * dis[dst[e]];
}

// H[N,64] = (RELU? max(X,0):X)[N,64] @ W[64,64]^T + b, via v_wmma_f32_16x16x32_f16.
// One wave handles a 16-row strip; 4 column tiles x 2 K-steps = 8 WMMAs per wave.
template<bool RELU>
__global__ __launch_bounds__(THREADS)
void k_gemm_wmma(const float* __restrict__ X, const float* __restrict__ W,
                 const float* __restrict__ bias, float* __restrict__ H, int N)
{
  const int wave = blockIdx.x * (THREADS / 32) + (threadIdx.x >> 5);
  const int r0 = wave * 16;
  if (r0 >= N) return;                          // wave-uniform: EXEC stays all-1s
  const bool full = (r0 + 16 <= N);             // wave-uniform full-tile test
  const int lane = threadIdx.x & 31;
  const int half = lane >> 4;
  const int l16  = lane & 15;

  // ---- A fragments: rows r0..r0+15, K=0..31 (a0) and K=32..63 (a1) ----
  v16h a0, a1;
  {
    int row = r0 + l16; if (row >= N) row = N - 1;
    const float* xr = X + (long long)row * 64;
#pragma unroll
    for (int p = 0; p < 8; ++p) {
      int k = kmapA(p, half);
      float2 v0 = *(const float2*)(xr + k);
      float2 v1 = *(const float2*)(xr + k + 32);
      if (RELU) {
        v0.x = fmaxf(v0.x, 0.f); v0.y = fmaxf(v0.y, 0.f);
        v1.x = fmaxf(v1.x, 0.f); v1.y = fmaxf(v1.y, 0.f);
      }
      a0[2*p] = (_Float16)v0.x; a0[2*p+1] = (_Float16)v0.y;
      a1[2*p] = (_Float16)v1.x; a1[2*p+1] = (_Float16)v1.y;
    }
  }

  // ---- 4 column tiles of 16 ----
#pragma unroll
  for (int t = 0; t < 4; ++t) {
    const int j = t * 16 + l16;                 // output column == W row
    const float* wr = W + (long long)j * 64;    // B[k][j] = W[j][k]
    v16h b0, b1;
#pragma unroll
    for (int p = 0; p < 8; ++p) {
      int k = kmapB(p, half);
      float2 v0 = *(const float2*)(wr + k);
      float2 v1 = *(const float2*)(wr + k + 32);
      b0[2*p] = (_Float16)v0.x; b0[2*p+1] = (_Float16)v0.y;
      b1[2*p] = (_Float16)v1.x; b1[2*p+1] = (_Float16)v1.y;
    }
    const float bv = bias[j];
    v8f c;
#pragma unroll
    for (int r = 0; r < 8; ++r) c[r] = bv;      // bias as accumulator init
    c = __builtin_amdgcn_wmma_f32_16x16x32_f16(false, a0, false, b0, (short)0, c, false, false);
    c = __builtin_amdgcn_wmma_f32_16x16x32_f16(false, a1, false, b1, (short)0, c, false, false);

    // C/D layout: VGPR r <-> row r0 + r + 8*half, lane%16 <-> column j.
    if (full) {                                 // wave-uniform: no exec juggling
      float* hp = H + ((long long)(r0 + 8 * half)) * 64 + j;
#pragma unroll
      for (int r = 0; r < 8; ++r) hp[(long long)r * 64] = c[r];
    } else {
#pragma unroll
      for (int r = 0; r < 8; ++r) {
        int row = r0 + r + 8 * half;
        if (row < N) H[(long long)row * 64 + j] = c[r];
      }
    }
  }
}

// out[i] = dis[i]^2 * (h[i] + ebond[4] + edir[0])  -- self-loop term, non-atomic init.
// 16 lanes per node, float4 per lane.
__global__ __launch_bounds__(THREADS)
void k_self_init(const float* __restrict__ H, const float* __restrict__ dis,
                 const float* __restrict__ ebond, const float* __restrict__ edir,
                 float* __restrict__ out, int N)
{
  int tid = blockIdx.x * THREADS + threadIdx.x;
  int i = tid >> 4;
  if (i >= N) return;
  int c = (tid & 15) * 4;
  float di = dis[i];
  float nm = di * di;                           // dis[i]*dis[i] = 1/deg
  float4 hv = *(const float4*)(H + (long long)i * 64 + c);
  float4 eb = *(const float4*)(ebond + 4 * 64 + c);   // bond type 4 (self loop)
  float4 ed = *(const float4*)(edir + c);             // dir 0
  float4 o;
  o.x = nm * (hv.x + eb.x + ed.x);
  o.y = nm * (hv.y + eb.y + ed.y);
  o.z = nm * (hv.z + eb.z + ed.z);
  o.w = nm * (hv.w + eb.w + ed.w);
  *(float4*)(out + (long long)i * 64 + c) = o;
}

// out[dst] += norm * (h[src] + ebond[bt] + edir[bd]); 16 lanes/edge, float4/lane.
__global__ __launch_bounds__(THREADS)
void k_edge_scatter(const float* __restrict__ H, const float* __restrict__ norm,
                    const int* __restrict__ src, const int* __restrict__ dst,
                    const int* __restrict__ attr, const float* __restrict__ ebond,
                    const float* __restrict__ edir, float* __restrict__ out, int E)
{
  int tid = blockIdx.x * THREADS + threadIdx.x;
  int e = tid >> 4;
  if (e >= E) return;
  int c = (tid & 15) * 4;
  int s = src[e], d = dst[e];
  float nm = norm[e];
  int bt = attr[2 * e], bd = attr[2 * e + 1];
  float4 hv = *(const float4*)(H + (long long)s * 64 + c);
  float4 eb = *(const float4*)(ebond + bt * 64 + c);
  float4 ed = *(const float4*)(edir + bd * 64 + c);
  float* o = out + (long long)d * 64 + c;
  atomicAdd(o + 0, nm * (hv.x + eb.x + ed.x));
  atomicAdd(o + 1, nm * (hv.y + eb.y + ed.y));
  atomicAdd(o + 2, nm * (hv.z + eb.z + ed.z));
  atomicAdd(o + 3, nm * (hv.w + eb.w + ed.w));
}

extern "C" void kernel_launch(void* const* d_in, const int* in_sizes, int n_in,
                              void* d_out, int out_size, void* d_ws, size_t ws_size,
                              hipStream_t stream) {
  const float* x    = (const float*)d_in[0];
  const int*   eidx = (const int*)  d_in[1];   // [2, E]
  const int*   attr = (const int*)  d_in[2];   // [E, 2]
  const float* W1   = (const float*)d_in[3];
  const float* b1   = (const float*)d_in[4];
  const float* eb1  = (const float*)d_in[5];
  const float* ed1  = (const float*)d_in[6];
  const float* W2   = (const float*)d_in[7];
  const float* b2   = (const float*)d_in[8];
  const float* eb2  = (const float*)d_in[9];
  const float* ed2  = (const float*)d_in[10];
  float* out = (float*)d_out;

  const int N = in_sizes[0] / 64;
  const int E = in_sizes[1] / 2;
  const int* src = eidx;
  const int* dst = eidx + E;

  float* ws   = (float*)d_ws;
  float* dis  = ws;                               // [N]
  float* nrm  = dis + N;                          // [E]
  float* h    = nrm + E;                          // [N,64]
  float* out1 = h + (long long)N * 64;            // [N,64]

  const int gN   = (N + THREADS - 1) / THREADS;
  const int gE   = (E + THREADS - 1) / THREADS;
  const int gN16 = (int)(((long long)N * 16 + THREADS - 1) / THREADS);
  const int gE16 = (int)(((long long)E * 16 + THREADS - 1) / THREADS);
  const int gGemm = ((N + 15) / 16 + (THREADS / 32) - 1) / (THREADS / 32);

  // degree -> dis -> per-edge norm (shared by both layers)
  k_deg_init <<<gN, THREADS, 0, stream>>>(dis, N);
  k_deg_count<<<gE, THREADS, 0, stream>>>(src, dis, E);
  k_dis      <<<gN, THREADS, 0, stream>>>(dis, N);
  k_norm     <<<gE, THREADS, 0, stream>>>(src, dst, dis, nrm, E);

  // layer 1
  k_gemm_wmma<false><<<gGemm, THREADS, 0, stream>>>(x, W1, b1, h, N);
  k_self_init      <<<gN16, THREADS, 0, stream>>>(h, dis, eb1, ed1, out1, N);
  k_edge_scatter   <<<gE16, THREADS, 0, stream>>>(h, nrm, src, dst, attr, eb1, ed1, out1, E);

  // layer 2 (ReLU fused into the GEMM's A load)
  k_gemm_wmma<true> <<<gGemm, THREADS, 0, stream>>>(out1, W2, b2, h, N);
  k_self_init      <<<gN16, THREADS, 0, stream>>>(h, dis, eb2, ed2, out, N);
  k_edge_scatter   <<<gE16, THREADS, 0, stream>>>(h, nrm, src, dst, attr, eb2, ed2, out, E);
}
